// RWKV6Attention_85117661872239
// MI455X (gfx1250) — matrix operand order repaired and checked by reading, not compiled, not run on hardware
//
#include <hip/hip_runtime.h>
#include <hip/hip_bf16.h>
#include <cstddef>

// ---------------- types & helpers ----------------
typedef __attribute__((ext_vector_type(16))) __bf16 v16bf;
typedef __attribute__((ext_vector_type(8)))  float  v8f;

static __device__ inline v8f wmma_bf(v16bf a, v16bf b, v8f c) {
    return __builtin_amdgcn_wmma_f32_16x16x32_bf16(false, a, false, b, (short)0, c, false, false);
}

// A-operand inverse mapping (contiguous k order, k = 16*half + elem):
// given contraction index k (0..31) -> (half, elem)
static __device__ inline int a_half(int k) { return (k >> 4) & 1; }
static __device__ inline int a_elem(int k) {
    return k & 15;
}
static __device__ inline unsigned pack_bf16(float f0, float f1) {
    __bf16 b0 = (__bf16)f0, b1 = (__bf16)f1;
    return (unsigned)__builtin_bit_cast(unsigned short, b0) |
           ((unsigned)__builtin_bit_cast(unsigned short, b1) << 16);
}

#define T_TOK 4096
#define HID   1024
#define KD    512
#define VD    1024
#define NH    4
#define DK    128
#define DV    256
#define LRK   32
#define NCH   64

// ---------------- kernel 1: z0 = x + (shift(x)-x)*mu ----------------
__global__ __launch_bounds__(256) void lerp0_kernel(const float* __restrict__ x,
                                                    const float* __restrict__ mu,
                                                    __bf16* __restrict__ z0) {
    int idx = blockIdx.x * 256 + threadIdx.x;
    int t = idx >> 10, h = idx & 1023;
    int l = t & 2047;
    float xv = x[(size_t)t * HID + h];
    float xp = (l == 0) ? 0.f : x[(size_t)(t - 1) * HID + h];
    z0[(size_t)t * HID + h] = (__bf16)(xv + (xp - xv) * mu[h]);
}

// ---------------- kernel 2: generic WMMA bf16 GEMM: C = act(A @ W^T + bias) ----------------
// A: [M,K] bf16 row-major. W: [N,K] fp32 row-major. block tile 128(M) x 64(N).
// Staging: vector global loads -> fragment-layout LDS; MAC loop is branch-free
// (out-of-range N columns staged as zeros); only the epilogue guards stores.
__global__ __launch_bounds__(256) void gemm_kernel(const __bf16* __restrict__ A,
                                                   const float* __restrict__ W,
                                                   const float* __restrict__ bias,
                                                   float* __restrict__ outF,
                                                   __bf16* __restrict__ outB,
                                                   int M, int N, int K, int act) {
    __shared__ __attribute__((aligned(32))) __bf16 AF[8 * 32 * 16]; // 8 m-tiles
    __shared__ __attribute__((aligned(32))) __bf16 BF[4 * 32 * 16]; // 4 n-tiles

    int tid = threadIdx.x;
    int lane = tid & 31, wave = tid >> 5;
    int half = lane >> 4, l15 = lane & 15;
    int mbase_blk = blockIdx.x * 128;
    int nbase0 = blockIdx.y * 64;

    // staging geometry (single pass, 256 threads)
    const int ai  = tid >> 1;              // A row 0..127
    const int akk0 = (tid & 1) * 16;       // A k-group base
    const int bn  = tid >> 2;              // B row 0..63
    const int bkk0 = (tid & 3) * 8;        // B k-group base

    v8f z = {0.f, 0.f, 0.f, 0.f, 0.f, 0.f, 0.f, 0.f};
    v8f acc0 = z, acc1 = z, acc2 = z, acc3 = z;

    for (int k = 0; k < K; k += 32) {
        __syncthreads();
        {   // stage A: 16 consecutive bf16 per thread (2x b128 loads, 8x b32 LDS stores)
            const uint4* src = (const uint4*)(A + (size_t)(mbase_blk + ai) * K + k + akk0);
            uint4 w0 = src[0], w1 = src[1];
            unsigned words[8] = {w0.x, w0.y, w0.z, w0.w, w1.x, w1.y, w1.z, w1.w};
            int mt = ai >> 4;
#pragma unroll
            for (int j = 0; j < 8; j++) {
                int kk = akk0 + 2 * j;
                int ln = (ai & 15) + (a_half(kk) << 4);
                *(unsigned*)&AF[(((mt * 32) + ln) << 4) + a_elem(kk)] = words[j];
            }
        }
        {   // stage B: 8 consecutive fp32 per thread -> packed bf16, one b128 LDS store
            int nb = nbase0 + bn;
            float f[8];
            if (nb < N) {
                const float4* src = (const float4*)(W + (size_t)nb * K + k + bkk0);
                float4 a0 = src[0], a1 = src[1];
                f[0] = a0.x; f[1] = a0.y; f[2] = a0.z; f[3] = a0.w;
                f[4] = a1.x; f[5] = a1.y; f[6] = a1.z; f[7] = a1.w;
            } else {
#pragma unroll
                for (int j = 0; j < 8; j++) f[j] = 0.f;
            }
            int nt = bn >> 4;
            int ln = (bn & 15) + ((bkk0 >> 4) << 4);
            uint4 wds;
            wds.x = pack_bf16(f[0], f[1]);
            wds.y = pack_bf16(f[2], f[3]);
            wds.z = pack_bf16(f[4], f[5]);
            wds.w = pack_bf16(f[6], f[7]);
            *(uint4*)&BF[(((nt * 32) + ln) << 4) + (bkk0 & 15)] = wds;
        }
        __syncthreads();
        v16bf a  = *(const v16bf*)&AF[(wave * 32 + lane) << 4];
        v16bf b0 = *(const v16bf*)&BF[(0 * 32 + lane) << 4];
        v16bf b1 = *(const v16bf*)&BF[(1 * 32 + lane) << 4];
        v16bf b2 = *(const v16bf*)&BF[(2 * 32 + lane) << 4];
        v16bf b3 = *(const v16bf*)&BF[(3 * 32 + lane) << 4];
        acc0 = wmma_bf(a, b0, acc0);
        acc1 = wmma_bf(a, b1, acc1);
        acc2 = wmma_bf(a, b2, acc2);
        acc3 = wmma_bf(a, b3, acc3);
    }
    v8f accs[4] = {acc0, acc1, acc2, acc3};
#pragma unroll
    for (int j = 0; j < 4; j++) {
        int nb = nbase0 + j * 16;
        if (nb >= N) continue;
        int col = nb + l15;
#pragma unroll
        for (int r = 0; r < 8; r++) {
            int row = mbase_blk + wave * 16 + r + half * 8;
            float vv = accs[j][r];
            if (bias) vv += bias[col];
            if (act) vv = tanhf(vv);
            if (outF) outF[(size_t)row * N + col] = vv;
            if (outB) outB[(size_t)row * N + col] = (__bf16)vv;
        }
    }
}

// ---------------- kernel 3: mix einsum + five lerped bf16 activations ----------------
__global__ __launch_bounds__(256) void mix_lerp_kernel(const float* __restrict__ x,
                                                       const float* __restrict__ x5,
                                                       const float* __restrict__ x2w,
                                                       const float* __restrict__ xbias,
                                                       __bf16* __restrict__ Xall) {
    int t = blockIdx.x, tid = threadIdx.x;
    __shared__ float s5[160];
    if (tid < 160) s5[tid] = x5[(size_t)t * 160 + tid];
    __syncthreads();
    int l = t & 2047;
    for (int h = tid; h < HID; h += 256) {
        float xv = x[(size_t)t * HID + h];
        float xp = (l == 0) ? 0.f : x[(size_t)(t - 1) * HID + h];
        float d = xp - xv;
#pragma unroll
        for (int i = 0; i < 5; i++) {
            float m = xbias[i * HID + h];
            const float* wrow = x2w + (size_t)h * 160 + i * LRK;
#pragma unroll
            for (int r = 0; r < LRK; r++) m += s5[i * LRK + r] * wrow[r];
            Xall[(size_t)i * T_TOK * HID + (size_t)t * HID + h] = (__bf16)(xv + d * m);
        }
    }
}

// ---------------- kernel 4: chunked RWKV6 recurrence (state in LDS) ----------------
// State S[DK][DV] kept transposed (ST[p][d]) in LDS for the whole sequence.
// qe / kw / A_intra stored pre-swizzled in A-fragment layout; v stored transposed;
// ke row-major (its B-operand reads are contiguous along d).
__global__ __launch_bounds__(256) void rwkv_attn_kernel(const float* __restrict__ rb,
                                                        const float* __restrict__ wb,
                                                        const float* __restrict__ kb,
                                                        const float* __restrict__ vb,
                                                        const float* __restrict__ bonus,
                                                        float* __restrict__ ob) {
    int bh = blockIdx.x;
    int b = bh >> 2, h = bh & 3;
    int tid = threadIdx.x, lane = tid & 31, wave = tid >> 5;
    int half = lane >> 4, l15 = lane & 15;

    __shared__ float  wcs[32][DK];                                    // 16 KB
    __shared__ float  decay[DK];
    __shared__ __attribute__((aligned(32))) __bf16 qeF[8 * 32 * 16];  // 8 KB  (2 mt x 4 kt)
    __shared__ __attribute__((aligned(32))) __bf16 keR[32][DK];       // 8 KB  row-major
    __shared__ __attribute__((aligned(32))) __bf16 kwF[8 * 32 * 16];  // 8 KB  (8 dt)
    __shared__ __attribute__((aligned(32))) __bf16 vT[DV][32];        // 16 KB transposed
    __shared__ __attribute__((aligned(32))) __bf16 ST[DV][DK];        // 64 KB state^T
    __shared__ __attribute__((aligned(32))) __bf16 AF2[2 * 32 * 16];  // 2 KB
    __shared__ float  diag[32];

    for (int i = tid; i < DV * DK; i += 256) ((__bf16*)ST)[i] = (__bf16)0.f;
    __syncthreads();

    const size_t base_k = (size_t)b * 2048 * KD + (size_t)h * DK;
    const size_t base_v = (size_t)b * 2048 * VD + (size_t)h * DV;
    v8f zacc = {0.f, 0.f, 0.f, 0.f, 0.f, 0.f, 0.f, 0.f};

    for (int c = 0; c < NCH; c++) {
        int l0 = c * 32;
        // prefetch next chunk's rows into L2 (global_prefetch_b8)
        if (c + 1 < NCH && tid < 32) {
            size_t nb = base_k + (size_t)(l0 + 32 + tid) * KD;
            __builtin_prefetch(&rb[nb], 0, 1);
            __builtin_prefetch(&wb[nb], 0, 1);
            __builtin_prefetch(&kb[nb], 0, 1);
            __builtin_prefetch(&vb[base_v + (size_t)(l0 + 32 + tid) * VD], 0, 1);
        }
        // ---- cumsum of w = -exp(w_raw) along time, per d column ----
        if (tid < DK) {
            int d = tid;
            float acc = 0.f;
            for (int i = 0; i < 32; i++) {
                acc += -expf(wb[base_k + (size_t)(l0 + i) * KD + d]);
                wcs[i][d] = acc;
            }
            decay[d] = expf(acc);
        }
        __syncthreads();
        // ---- build qe (A-frag), ke (row-major), kw (A-frag over time) ----
        for (int idx = tid; idx < 32 * DK; idx += 256) {
            int i = idx >> 7, d = idx & (DK - 1);
            float wlast = wcs[31][d], wc = wcs[i][d];
            float wv = -expf(wb[base_k + (size_t)(l0 + i) * KD + d]);
            float q = rb[base_k + (size_t)(l0 + i) * KD + d];
            float k = kb[base_k + (size_t)(l0 + i) * KD + d];
            {   // qe fragment: mt = i>>4, kt = d>>5, contraction index kk = d&31
                int mt = i >> 4, kt = d >> 5, kk = d & 31;
                int ln = (i & 15) + (a_half(kk) << 4);
                qeF[((((mt * 4 + kt) * 32) + ln) << 4) + a_elem(kk)] =
                    (__bf16)(q * expf(wc - wv));
            }
            keR[i][d] = (__bf16)(k * expf(-wc));
            {   // kw^T fragment: dt = d>>4, rows are d, contraction index is time i
                int dt = d >> 4;
                int ln = (d & 15) + (a_half(i) << 4);
                kwF[(((dt * 32) + ln) << 4) + a_elem(i)] =
                    (__bf16)(k * expf(wlast - wc));
            }
        }
        for (int idx = tid; idx < 32 * DV; idx += 256) {
            int i = idx >> 8, p = idx & (DV - 1);
            vT[p][i] = (__bf16)vb[base_v + (size_t)(l0 + i) * VD + p];
        }
        if (tid < 32) {
            int i = tid;
            float s = 0.f;
            for (int d = 0; d < DK; d++)
                s += rb[base_k + (size_t)(l0 + i) * KD + d] * bonus[h * DK + d] *
                     kb[base_k + (size_t)(l0 + i) * KD + d];
            diag[i] = s;
        }
        __syncthreads();

        // ---- A = qe @ ke^T (waves 0..3), mask j<i, write to A-frag layout ----
        if (wave < 4) {
            int mt = wave >> 1, jt = wave & 1;
            v8f acc = zacc;
            int j = jt * 16 + l15;
#pragma unroll
            for (int kt = 0; kt < 4; kt++) {
                v16bf a = *(const v16bf*)&qeF[((mt * 4 + kt) * 32 + lane) << 4];
                v16bf bm = *(const v16bf*)&keR[j][kt * 32 + half * 16];
                acc = wmma_bf(a, bm, acc);
            }
#pragma unroll
            for (int r = 0; r < 8; r++) {
                int i = mt * 16 + r + half * 8;
                float v = (j < i) ? acc[r] : 0.f;
                int ln = (i & 15) + (a_half(j) << 4);
                AF2[(((mt * 32) + ln) << 4) + a_elem(j)] = (__bf16)v;
            }
        }
        __syncthreads();

        // ---- o = qe @ S + A @ v + diag * v  (32 tiles, 4 per wave) ----
        for (int j4 = 0; j4 < 4; j4++) {
            int id = wave * 4 + j4;
            int mt = id >> 4, nt = id & 15;
            int p = nt * 16 + l15;
            v8f acc = zacc;
#pragma unroll
            for (int kt = 0; kt < 4; kt++) {
                v16bf a = *(const v16bf*)&qeF[((mt * 4 + kt) * 32 + lane) << 4];
                v16bf bm = *(const v16bf*)&ST[p][kt * 32 + half * 16];
                acc = wmma_bf(a, bm, acc);
            }
            {   // intra-chunk: A[32x32] @ v[32xDV]
                v16bf a2 = *(const v16bf*)&AF2[(mt * 32 + lane) << 4];
                v16bf b2 = *(const v16bf*)&vT[p][half * 16];
                acc = wmma_bf(a2, b2, acc);
            }
#pragma unroll
            for (int r = 0; r < 8; r++) {
                int i = mt * 16 + r + half * 8;
                float o = acc[r] + diag[i] * (float)vT[p][i];
                ob[(size_t)(b * 2048 + l0 + i) * (NH * DV) + (size_t)h * DV + p] = o;
            }
        }
        __syncthreads();

        // ---- state update: S = S*decay + kw^T @ v (wave -> its 16-row d strip) ----
        {
            int dt = wave;
            v16bf a = *(const v16bf*)&kwF[(dt * 32 + lane) << 4];
            for (int nt = 0; nt < 16; nt++) {
                int p = nt * 16 + l15;
                v16bf bm = *(const v16bf*)&vT[p][half * 16];
                v8f cc;
#pragma unroll
                for (int r = 0; r < 8; r++) {
                    int d = dt * 16 + r + half * 8;
                    cc[r] = (float)ST[p][d] * decay[d];
                }
                cc = wmma_bf(a, bm, cc);
#pragma unroll
                for (int r = 0; r < 8; r++) {
                    int d = dt * 16 + r + half * 8;
                    ST[p][d] = (__bf16)cc[r];
                }
            }
        }
        __syncthreads();
    }
}

// ---------------- kernel 5: GroupNorm(NH) + swish gate -> bf16 ----------------
__global__ __launch_bounds__(256) void gn_gate_kernel(const float* __restrict__ o,
                                                      const float* __restrict__ g,
                                                      const float* __restrict__ gnw,
                                                      const float* __restrict__ gnb,
                                                      __bf16* __restrict__ gated) {
    int t = blockIdx.x, tid = threadIdx.x;
    __shared__ float rs[256], rq[256];
    for (int hd = 0; hd < NH; hd++) {
        int c = hd * DV + tid;
        float v = o[(size_t)t * VD + c];
        rs[tid] = v;
        rq[tid] = v * v;
        __syncthreads();
        for (int s = 128; s > 0; s >>= 1) {
            if (tid < s) { rs[tid] += rs[tid + s]; rq[tid] += rq[tid + s]; }
            __syncthreads();
        }
        float mean = rs[0] * (1.f / 256.f);
        float var  = rq[0] * (1.f / 256.f) - mean * mean;
        __syncthreads();
        float on = (v - mean) * rsqrtf(var + 1e-5f) * gnw[c] + gnb[c];
        float gv = g[(size_t)t * VD + c];
        float sw = gv / (1.f + expf(-gv));
        gated[(size_t)t * VD + c] = (__bf16)(on * sw);
    }
}

// ---------------- host launcher ----------------
extern "C" void kernel_launch(void* const* d_in, const int* in_sizes, int n_in,
                              void* d_out, int out_size, void* d_ws, size_t ws_size,
                              hipStream_t stream) {
    (void)in_sizes; (void)n_in; (void)out_size; (void)ws_size;
    const float* x     = (const float*)d_in[0];
    const float* x0mu  = (const float*)d_in[1];
    const float* x0w   = (const float*)d_in[2];
    const float* x2w   = (const float*)d_in[3];
    const float* xbias = (const float*)d_in[4];
    const float* rw    = (const float*)d_in[5];
    const float* ww1   = (const float*)d_in[6];
    const float* ww2   = (const float*)d_in[7];
    const float* wb2   = (const float*)d_in[8];
    const float* kw    = (const float*)d_in[9];
    const float* vw    = (const float*)d_in[10];
    const float* gw    = (const float*)d_in[11];
    const float* bonus = (const float*)d_in[12];
    const float* gnw   = (const float*)d_in[13];
    const float* gnb   = (const float*)d_in[14];
    const float* ow    = (const float*)d_in[15];

    char* p = (char*)d_ws;
    auto alloc = [&](size_t bytes) -> void* {
        void* r = (void*)p;
        p += (bytes + 255) & ~(size_t)255;
        return r;
    };
    __bf16* z0    = (__bf16*)alloc((size_t)T_TOK * HID * 2);
    float*  x5    = (float*) alloc((size_t)T_TOK * 160 * 4);
    __bf16* Xall  = (__bf16*)alloc((size_t)5 * T_TOK * HID * 2);
    __bf16* wtmp  = (__bf16*)alloc((size_t)T_TOK * 64 * 2);
    float*  rbuf  = (float*) alloc((size_t)T_TOK * KD * 4);
    float*  wbuf  = (float*) alloc((size_t)T_TOK * KD * 4);
    float*  kbuf  = (float*) alloc((size_t)T_TOK * KD * 4);
    float*  vbuf  = (float*) alloc((size_t)T_TOK * VD * 4);
    float*  gbuf  = (float*) alloc((size_t)T_TOK * VD * 4);
    float*  obuf  = (float*) alloc((size_t)T_TOK * VD * 4);
    __bf16* gated = (__bf16*)alloc((size_t)T_TOK * VD * 2);

    const int MB = T_TOK / 128; // 32 M-blocks

    lerp0_kernel<<<T_TOK * HID / 256, 256, 0, stream>>>(x, x0mu, z0);
    gemm_kernel<<<dim3(MB, 3), 256, 0, stream>>>(z0, x0w, nullptr, x5, nullptr,
                                                 T_TOK, 160, HID, 1);
    mix_lerp_kernel<<<T_TOK, 256, 0, stream>>>(x, x5, x2w, xbias, Xall);

    const size_t XS = (size_t)T_TOK * HID;
    gemm_kernel<<<dim3(MB, 8), 256, 0, stream>>>(Xall + 0 * XS, rw, nullptr, rbuf, nullptr,
                                                 T_TOK, KD, HID, 0);
    gemm_kernel<<<dim3(MB, 1), 256, 0, stream>>>(Xall + 1 * XS, ww1, nullptr, nullptr, wtmp,
                                                 T_TOK, 64, HID, 1);
    gemm_kernel<<<dim3(MB, 8), 256, 0, stream>>>(wtmp, ww2, wb2, wbuf, nullptr,
                                                 T_TOK, KD, 64, 0);
    gemm_kernel<<<dim3(MB, 8), 256, 0, stream>>>(Xall + 2 * XS, kw, nullptr, kbuf, nullptr,
                                                 T_TOK, KD, HID, 0);
    gemm_kernel<<<dim3(MB, 16), 256, 0, stream>>>(Xall + 3 * XS, vw, nullptr, vbuf, nullptr,
                                                  T_TOK, VD, HID, 0);
    gemm_kernel<<<dim3(MB, 16), 256, 0, stream>>>(Xall + 4 * XS, gw, nullptr, gbuf, nullptr,
                                                  T_TOK, VD, HID, 0);

    rwkv_attn_kernel<<<2 * NH, 256, 0, stream>>>(rbuf, wbuf, kbuf, vbuf, bonus, obuf);

    gn_gate_kernel<<<T_TOK, 256, 0, stream>>>(obuf, gbuf, gnw, gnb, gated);
    gemm_kernel<<<dim3(MB, 16), 256, 0, stream>>>(gated, ow, nullptr, (float*)d_out, nullptr,
                                                  T_TOK, VD, VD, 0);
}
